// LightGCN_12043088298585
// MI455X (gfx1250) — matrix operand order repaired and checked by reading, not compiled, not run on hardware
//
#include <hip/hip_runtime.h>
#include <hip/hip_bf16.h>
#include <hip/hip_fp16.h>
#include <math.h>

#define NUM_USERS 100000
#define NUM_ITEMS 50000
#define N_NODES   (NUM_USERS + NUM_ITEMS)
#define LATENT    64
#define N_LAYERS  3
#define N_EDGES   2400000
#define BATCH     16384

typedef __attribute__((ext_vector_type(16))) _Float16 v16h;
typedef __attribute__((ext_vector_type(8)))  float    v8f;

// ---- x0 = concat(user_emb, item_emb); acc = x0; zero first scatter buffer ----
__global__ void k_init(const float* __restrict__ ue, const float* __restrict__ ie,
                       float* __restrict__ x, float* __restrict__ acc,
                       float* __restrict__ zbuf) {
    size_t t = (size_t)blockIdx.x * blockDim.x + threadIdx.x;
    size_t flat = t * 4;
    if (flat >= (size_t)N_NODES * LATENT) return;
    size_t node = flat >> 6;       // LATENT == 64
    size_t c    = flat & 63;
    const float* src = (node < (size_t)NUM_USERS)
        ? (ue + node * LATENT + c)
        : (ie + (node - NUM_USERS) * LATENT + c);
    float4 v = *(const float4*)src;
    *(float4*)(x + flat)   = v;
    *(float4*)(acc + flat) = v;
    float4 z{0.f, 0.f, 0.f, 0.f};
    *(float4*)(zbuf + flat) = z;
}

// ---- fused: acc += x_next, and zero the buffer that becomes the next target ----
__global__ void k_acczero(float* __restrict__ acc, const float* __restrict__ x,
                          float* __restrict__ zbuf, size_t n4) {
    size_t t = (size_t)blockIdx.x * blockDim.x + threadIdx.x;
    if (t >= n4) return;
    float4 a = ((const float4*)acc)[t];
    float4 b = ((const float4*)x)[t];
    a.x += b.x; a.y += b.y; a.z += b.z; a.w += b.w;
    ((float4*)acc)[t] = a;
    float4 z{0.f, 0.f, 0.f, 0.f};
    ((float4*)zbuf)[t] = z;
}

// -------- edge-parallel SpMM scatter: one wave per edge, float2 per lane --------
// The 38.4MB feature table is L2-resident (192MB L2); this kernel is bound by
// L2 gather + native f32 atomic-add throughput. Edge metadata is wave-uniform,
// so force it scalar (s_load + broadcast) via readfirstlane.
__global__ void k_spmm(const int* __restrict__ rows, const int* __restrict__ cols,
                       const float* __restrict__ vals,
                       const float* __restrict__ xin, float* __restrict__ xout) {
    int wave_in_blk = __builtin_amdgcn_readfirstlane((int)(threadIdx.x >> 5));
    size_t e = (size_t)blockIdx.x * 8 + (size_t)wave_in_blk;   // 256 thr = 8 waves
    if (e >= (size_t)N_EDGES) return;
    int lane = (int)(threadIdx.x & 31);
    int c = lane * 2;
    int   r  = rows[e];     // wave-uniform -> scalar loads
    int   cl = cols[e];
    float v  = vals[e];
    float2 x = *(const float2*)(xin + (size_t)cl * LATENT + c);
    float* dst = xout + (size_t)r * LATENT + c;
    unsafeAtomicAdd(dst,     v * x.x);   // guaranteed global_atomic_add_f32
    unsafeAtomicAdd(dst + 1, v * x.y);
}

// -------- fused gather + MLP, one wave per 16 batch rows, WMMA f16->f32 --------
__global__ void __launch_bounds__(32)
k_mlp(const int* __restrict__ users, const int* __restrict__ items,
      const float* __restrict__ acc,
      const float* __restrict__ W0, const float* __restrict__ b0,
      const float* __restrict__ W1, const float* __restrict__ b1,
      const float* __restrict__ Wa, const float* __restrict__ ba,
      float* __restrict__ out) {
    __shared__ _Float16 Vt[16][128];   // concat(u,i)/4 in f16
    __shared__ _Float16 H0[16][64];    // relu(V@W0+b0)
    __shared__ float    H1[16][32];    // relu(H0@W1+b1)

    const int lane = threadIdx.x;
    const int base = blockIdx.x * 16;
    const int g  = lane >> 4;          // half-wave group (K-group for WMMA layouts)
    const int nl = lane & 15;

    // ---- gather 16 rows: even lanes -> user half, odd lanes -> item half ----
    {
        int r    = lane >> 1;
        int half = lane & 1;
        size_t node = half ? (size_t)NUM_USERS + (size_t)items[base + r]
                           : (size_t)users[base + r];
        const float* src = acc + node * LATENT;
        #pragma unroll
        for (int c = 0; c < 64; ++c)
            Vt[r][half * 64 + c] = (_Float16)(src[c] * 0.25f);   // /(N_LAYERS+1)
    }
    __syncthreads();

    // ---- layer 0: [16x128] @ W0[128x64] -> H0[16x64] (4 N-tiles x 4 K-chunks) ----
    #pragma unroll
    for (int n = 0; n < 4; ++n) {
        v8f cacc = {};
        #pragma unroll
        for (int kc = 0; kc < 4; ++kc) {
            v16h a, b;
            #pragma unroll
            for (int e = 0; e < 16; ++e) {
                int ka = kc * 32 + ((e < 8) ? 0 : 16) + g * 8 + (e & 7); // A 16x32 f16 layout
                a[e] = Vt[nl][ka];
                int kb = kc * 32 + ((e < 8) ? 0 : 16) + g * 8 + (e & 7); // B 32x16 f16 layout
                b[e] = (_Float16)W0[kb * 64 + n * 16 + nl];
            }
            cacc = __builtin_amdgcn_wmma_f32_16x16x32_f16(
                false, a, false, b, (short)0, cacc, false, false);
        }
        #pragma unroll
        for (int r = 0; r < 8; ++r) {
            int row = g * 8 + r;                    // C/D layout: M = r + 8*group
            int col = n * 16 + nl;
            float hv = cacc[r] + b0[col];
            H0[row][col] = (_Float16)(hv > 0.f ? hv : 0.f);
        }
    }
    __syncthreads();

    // ---- layer 1: [16x64] @ W1[64x32] -> H1[16x32] (2 N-tiles x 2 K-chunks) ----
    #pragma unroll
    for (int n = 0; n < 2; ++n) {
        v8f cacc = {};
        #pragma unroll
        for (int kc = 0; kc < 2; ++kc) {
            v16h a, b;
            #pragma unroll
            for (int e = 0; e < 16; ++e) {
                int ka = kc * 32 + ((e < 8) ? 0 : 16) + g * 8 + (e & 7);
                a[e] = H0[nl][ka];
                int kb = kc * 32 + ((e < 8) ? 0 : 16) + g * 8 + (e & 7);
                b[e] = (_Float16)W1[kb * 32 + n * 16 + nl];
            }
            cacc = __builtin_amdgcn_wmma_f32_16x16x32_f16(
                false, a, false, b, (short)0, cacc, false, false);
        }
        #pragma unroll
        for (int r = 0; r < 8; ++r) {
            int row = g * 8 + r;
            int col = n * 16 + nl;
            float hv = cacc[r] + b1[col];
            H1[row][col] = hv > 0.f ? hv : 0.f;
        }
    }
    __syncthreads();

    // ---- layer 2: logits + sigmoid (VALU; lanes 0..15 only) ----
    if (lane < 16) {
        float s = ba[0];
        #pragma unroll
        for (int j = 0; j < 32; ++j) s += H1[lane][j] * Wa[j];
        out[base + lane] = 1.f / (1.f + expf(-s));
    }
}

extern "C" void kernel_launch(void* const* d_in, const int* in_sizes, int n_in,
                              void* d_out, int out_size, void* d_ws, size_t ws_size,
                              hipStream_t stream) {
    const int*   users = (const int*)d_in[0];
    const int*   items = (const int*)d_in[1];
    const int*   rows  = (const int*)d_in[2];
    const int*   cols  = (const int*)d_in[3];
    const float* vals  = (const float*)d_in[4];
    const float* ue    = (const float*)d_in[5];
    const float* ie    = (const float*)d_in[6];
    const float* W0    = (const float*)d_in[7];
    const float* b0    = (const float*)d_in[8];
    const float* W1    = (const float*)d_in[9];
    const float* b1    = (const float*)d_in[10];
    const float* Wa    = (const float*)d_in[11];
    const float* ba    = (const float*)d_in[12];
    float* out = (float*)d_out;
    (void)in_sizes; (void)n_in; (void)out_size; (void)ws_size;

    const size_t NN64 = (size_t)N_NODES * LATENT;   // 9.6M floats per buffer
    float* bufA = (float*)d_ws;                     // needs 3*38.4MB of workspace
    float* bufB = bufA + NN64;
    float* acc  = bufB + NN64;

    const size_t n4 = NN64 / 4;
    const int blk = 256;
    // x0 = concat; acc = x0; bufB zeroed (first scatter target)
    k_init<<<(unsigned)((n4 + blk - 1) / blk), blk, 0, stream>>>(ue, ie, bufA, acc, bufB);

    float* xin  = bufA;
    float* xout = bufB;
    for (int l = 0; l < N_LAYERS; ++l) {
        unsigned nblk_spmm = (unsigned)((N_EDGES + 7) / 8);    // 8 edges (waves) per block
        k_spmm<<<nblk_spmm, blk, 0, stream>>>(rows, cols, vals, xin, xout);
        // acc += xout; zero xin (it becomes the next scatter target)
        k_acczero<<<(unsigned)((n4 + blk - 1) / blk), blk, 0, stream>>>(acc, xout, xin, n4);
        float* t = xin; xin = xout; xout = t;
    }

    k_mlp<<<BATCH / 16, 32, 0, stream>>>(users, items, acc,
                                         W0, b0, W1, b1, Wa, ba, out);
}
